// TransformerBlock_49263274885990
// MI455X (gfx1250) — matrix operand order repaired and checked by reading, not compiled, not run on hardware
//
#include <hip/hip_runtime.h>
#include <math.h>

// ---------------- problem constants ----------------
static constexpr int Bc = 4, T = 2048, E = 1024, H = 16, Dh = 64;
static constexpr int MT = Bc * T;            // 8192 rows
static constexpr int E3 = 3 * E;             // 3072
static constexpr int E4 = 4 * E;             // 4096
static constexpr float EPS    = 1e-5f;
static constexpr float SCALE  = 0.022097086912079608f;   // 2048^-0.5

typedef __attribute__((ext_vector_type(16))) _Float16  v16h;
typedef __attribute__((ext_vector_type(8)))  float     v8f;
typedef __attribute__((ext_vector_type(4)))  unsigned  v4u;
typedef __attribute__((ext_vector_type(8)))  int       v8i;
typedef __attribute__((ext_vector_type(4)))  int       v4i;

#define WMMA_F16(a, b, c) \
  __builtin_amdgcn_wmma_f32_16x16x32_f16(false, (a), false, (b), (short)0, (c), false, false)

#if __has_builtin(__builtin_amdgcn_tensor_load_to_lds) && \
    __has_builtin(__builtin_amdgcn_s_wait_tensorcnt)
#define USE_TDM 1
#else
#define USE_TDM 0
#endif

// ---------------- WMMA fragment loaders (contiguous k order, same for A and B) ----------------
// A 16x32 f16 (MxK): lane m=lane%16, half=lane/16; VGPR v -> k = 16*half + 2v
__device__ inline v16h load_a_rm(const _Float16* base, int ld) {
  const int lane = threadIdx.x & 31;
  const int m = lane & 15, half = lane >> 4;
  const _Float16* p = base + m * ld + half * 16;
  v16h a;
#pragma unroll
  for (int v = 0; v < 8; ++v) {
    const int k = 2 * v;
    a[2 * v]     = p[k];
    a[2 * v + 1] = p[k + 1];
  }
  return a;
}

// B 32x16 f16 (KxN) sourced from a row-major [N x K] array (i.e. B = src^T):
// lane n=lane%16, half=lane/16; VGPR v -> k = 16*half + 2v  (contiguous pair in src row n)
__device__ inline v16h load_b_tn(const _Float16* base, int ld) {
  const int lane = threadIdx.x & 31;
  const int n = lane & 15, half = lane >> 4;
  const _Float16* p = base + n * ld + half * 16;
  v16h b;
#pragma unroll
  for (int v = 0; v < 8; ++v) {
    b[2 * v]     = p[2 * v];
    b[2 * v + 1] = p[2 * v + 1];
  }
  return b;
}

// B 32x16 f16 sourced from a row-major [K x N] array (strided along K).
__device__ inline v16h load_b_nn(const _Float16* base, int ld) {
  const int lane = threadIdx.x & 31;
  const int n = lane & 15, half = lane >> 4;
  v16h b;
#pragma unroll
  for (int v = 0; v < 8; ++v) {
    const int k = 16 * half + 2 * v;
    b[2 * v]     = base[k * ld + n];
    b[2 * v + 1] = base[(k + 1) * ld + n];
  }
  return b;
}

// ---------------- TDM: DMA a [64 rows x 32 halves] tile of BT into LDS ----------------
// D# per ISA 8.3/8.4: group0 = {count=1 | type=2, lds_addr, global_addr[56:0]};
// group1 = {data_size=2B, tensor_dim0=K, tensor_dim1=N, tile_dim0=32, tile_dim1=64,
//           tensor_dim0_stride=K}.  2D tensor -> groups 2/3 zero.
#if USE_TDM
__device__ inline void tdm_load_b_tile(unsigned lds_off, const _Float16* gsrc, int K, int N) {
  const unsigned long long ga = (unsigned long long)(size_t)gsrc;
  v4u g0;
  g0[0] = 1u;                                             // count=1 (valid user D#)
  g0[1] = lds_off;                                        // LDS byte address
  g0[2] = (unsigned)(ga & 0xffffffffull);                 // global_addr[95:64]
  g0[3] = (unsigned)((ga >> 32) & 0x01ffffffu) | (2u << 30); // addr[120:96] | type=2
  v8i g1;
  g1[0] = (int)(1u << 16);                                // data_size=1 (2 bytes)
  g1[1] = (int)(((unsigned)K & 0xffffu) << 16);           // tensor_dim0[15:0]
  g1[2] = (int)(((unsigned)K >> 16) | (((unsigned)N & 0xffffu) << 16)); // dim0 hi | dim1 lo
  g1[3] = (int)(((unsigned)N >> 16) | (32u << 16));       // dim1 hi | tile_dim0=32
  g1[4] = 64;                                             // tile_dim1=64 (tile_dim2=0)
  g1[5] = (int)(unsigned)K;                               // tensor_dim0_stride[31:0]
  g1[6] = 0;
  g1[7] = 0;
  v4i z = {0, 0, 0, 0};
#if defined(__clang_major__) && __clang_major__ >= 23
  v8i z8 = {};
  __builtin_amdgcn_tensor_load_to_lds(g0, g1, z, z, z8, 0);
#else
  __builtin_amdgcn_tensor_load_to_lds(g0, g1, z, z, 0);
#endif
}
#endif

// ---------------- weight fp32 -> f16 transpose:  wT[n,k] = w[k,n] ----------------
__global__ void k_cvt_transpose(const float* __restrict__ w, _Float16* __restrict__ wT,
                                int K, int N) {
  const size_t idx = (size_t)blockIdx.x * blockDim.x + threadIdx.x;
  if (idx >= (size_t)K * N) return;
  const int n = (int)(idx / K);
  const int k = (int)(idx % K);
  wT[idx] = (_Float16)w[(size_t)k * N + n];
}

// ---------------- LayerNorm (unbiased var, no bias) -> f16 ----------------
__global__ void k_layernorm_f16(const float* __restrict__ x, const float* __restrict__ g,
                                _Float16* __restrict__ out) {
  const int row  = blockIdx.x;
  const int tid  = threadIdx.x;
  const int lane = tid & 31, wave = tid >> 5;
  const float* xr = x + (size_t)row * E;

  float s = 0.f, s2 = 0.f;
#pragma unroll
  for (int i = 0; i < E / 256; ++i) {
    const float v = xr[tid + i * 256];
    s += v; s2 += v * v;
  }
#pragma unroll
  for (int off = 1; off < 32; off <<= 1) {
    s  += __shfl_xor(s,  off, 32);
    s2 += __shfl_xor(s2, off, 32);
  }
  __shared__ float a1[8], a2[8];
  __shared__ float mu_s, rstd_s;
  if (lane == 0) { a1[wave] = s; a2[wave] = s2; }
  __syncthreads();
  if (tid == 0) {
    float S = 0.f, S2 = 0.f;
#pragma unroll
    for (int w = 0; w < 8; ++w) { S += a1[w]; S2 += a2[w]; }
    const float mu  = S / (float)E;
    const float var = (S2 - (float)E * mu * mu) / (float)(E - 1);   // ddof=1
    mu_s   = mu;
    rstd_s = rsqrtf(var + EPS);
  }
  __syncthreads();
  const float mu = mu_s, rstd = rstd_s;
  _Float16* orow = out + (size_t)row * E;
#pragma unroll
  for (int i = 0; i < E / 256; ++i) {
    const int c = tid + i * 256;
    orow[c] = (_Float16)((xr[c] - mu) * rstd * g[c]);
  }
}

// ---------------- f16 WMMA GEMM:  C[MxN] = A[MxK] @ BT[NxK]^T ----------------
// Block = 4 waves, tile 128M x 64N; wave tile 32M x 64N (8 WMMAs / 32-K step).
// B tile (shared by all 4 waves) is staged into LDS by the Tensor Data Mover,
// ping-pong double buffered: TDM DMA for k+32 overlaps WMMAs for k.
__global__ void k_gemm_f16(const _Float16* __restrict__ A, const _Float16* __restrict__ BT,
                           const float* __restrict__ bias, const float* __restrict__ resid,
                           float* __restrict__ Cf32, _Float16* __restrict__ Cf16,
                           int M, int N, int K, int relu) {
  __shared__ _Float16 bstage[2][64 * 32];                 // 2 x 4 KiB
  const int lane = threadIdx.x & 31;
  const int wave = threadIdx.x >> 5;
  const int m0 = (blockIdx.y * 4 + wave) * 32;
  const int n0 = blockIdx.x * 64;
  const int n_ = lane & 15, half = lane >> 4;

  v8f acc[2][4] = {};
  const _Float16* pA = A + (size_t)m0 * K;

#if USE_TDM
  const unsigned lds0 = (unsigned)(size_t)&bstage[0][0];  // LDS aperture: offset in addr[31:0]
  if (wave == 0)
    tdm_load_b_tile(lds0, BT + (size_t)n0 * K, K, N);
  for (int k0 = 0, it = 0; k0 < K; k0 += 32, ++it) {
    const int p = it & 1;
    if (wave == 0) __builtin_amdgcn_s_wait_tensorcnt((short)0);
    __syncthreads();                                      // B tile p ready; tile 1-p free
    if (wave == 0 && k0 + 32 < K)
      tdm_load_b_tile(lds0 + (unsigned)((1 - p) * 64 * 32 * 2),
                      BT + (size_t)n0 * K + (k0 + 32), K, N);
    const v16h a0 = load_a_rm(pA + k0,                 K);
    const v16h a1 = load_a_rm(pA + (size_t)16 * K + k0, K);
#pragma unroll
    for (int t = 0; t < 4; ++t) {
      const v16h b = load_b_tn(&bstage[p][t * 16 * 32], 32);
      acc[0][t] = WMMA_F16(a0, b, acc[0][t]);
      acc[1][t] = WMMA_F16(a1, b, acc[1][t]);
    }
  }
#else
  for (int k0 = 0; k0 < K; k0 += 32) {
    __syncthreads();
    // cooperative fallback: 128 threads x 32 B = 4 KiB tile
#pragma unroll
    for (int j = 0; j < 4; ++j) {
      const int q   = 4 * ((int)threadIdx.x + 128 * j);   // half-index in tile
      const int row = q >> 5, col = q & 31;
      *(unsigned long long*)&bstage[0][q] =
          *(const unsigned long long*)&BT[(size_t)(n0 + row) * K + k0 + col];
    }
    __syncthreads();
    const v16h a0 = load_a_rm(pA + k0,                 K);
    const v16h a1 = load_a_rm(pA + (size_t)16 * K + k0, K);
#pragma unroll
    for (int t = 0; t < 4; ++t) {
      const v16h b = load_b_tn(&bstage[0][t * 16 * 32], 32);
      acc[0][t] = WMMA_F16(a0, b, acc[0][t]);
      acc[1][t] = WMMA_F16(a1, b, acc[1][t]);
    }
  }
#endif

#pragma unroll
  for (int u = 0; u < 2; ++u) {
#pragma unroll
    for (int t = 0; t < 4; ++t) {
#pragma unroll
      for (int r = 0; r < 8; ++r) {
        const int m   = r + 8 * half;
        const int row = m0 + u * 16 + m;
        const int col = n0 + 16 * t + n_;
        float v = acc[u][t][r];
        if (bias)  v += bias[col];
        if (relu)  v  = fmaxf(v, 0.f);
        if (resid) v += resid[(size_t)row * N + col];
        const size_t gi = (size_t)row * N + col;
        if (Cf16) Cf16[gi] = (_Float16)v;
        else      Cf32[gi] = v;
      }
    }
  }
}

// ---------------- attention helpers ----------------
// 16x16 score tile S = Q[i0:i0+16, :] @ K[j0:j0+16, :]^T  (D=64 -> 2 WMMAs)
__device__ inline v8f s_tile(const _Float16* Qb, const _Float16* Kb, int i0, int j0) {
  v8f c = {};
  v16h a0 = load_a_rm(Qb + (size_t)i0 * E3,      E3);
  v16h b0 = load_b_tn(Kb + (size_t)j0 * E3,      E3);
  c = WMMA_F16(a0, b0, c);
  v16h a1 = load_a_rm(Qb + (size_t)i0 * E3 + 32, E3);
  v16h b1 = load_b_tn(Kb + (size_t)j0 * E3 + 32, E3);
  c = WMMA_F16(a1, b1, c);
  return c;
}

// Pass 1: per-column (query-axis softmax!) running max & denominator.
__global__ void k_attn_colstats(const _Float16* __restrict__ qkv, float* __restrict__ cs) {
  const int j0 = blockIdx.x * 16, h = blockIdx.y, b = blockIdx.z;
  const int lane = threadIdx.x & 31, wave = threadIdx.x >> 5;
  const int n = lane & 15, half = lane >> 4;
  const _Float16* Qb = qkv + (size_t)b * T * E3 + h * Dh;
  const _Float16* Kb = Qb + E;

  float runm = -INFINITY, runs = 0.f;
  for (int it = (j0 >> 4) + wave; it < T / 16; it += 8) {
    const int i0 = it * 16;
    const v8f c = s_tile(Qb, Kb, i0, j0);
    float sv[8], tmax = -INFINITY;
#pragma unroll
    for (int r = 0; r < 8; ++r) {
      const int m = r + 8 * half;
      float s = c[r] * SCALE;
      s = ((i0 + m) >= (j0 + n)) ? s : -INFINITY;    // causal: need i >= j
      sv[r] = s;
      tmax = fmaxf(tmax, s);
    }
    tmax = fmaxf(tmax, __shfl_xor(tmax, 16, 32));
    const float newm = fmaxf(runm, tmax);
    float p = 0.f;
#pragma unroll
    for (int r = 0; r < 8; ++r) p += __expf(sv[r] - newm);
    p += __shfl_xor(p, 16, 32);
    runs = runs * __expf(runm - newm) + p;
    runm = newm;
  }
  __shared__ float smx[8][16], ssm[8][16];
  if (lane < 16) { smx[wave][lane] = runm; ssm[wave][lane] = runs; }
  __syncthreads();
  if (wave == 0 && lane < 16) {
    float M = -INFINITY;
#pragma unroll
    for (int w = 0; w < 8; ++w) M = fmaxf(M, smx[w][lane]);
    float Z = 0.f;
#pragma unroll
    for (int w = 0; w < 8; ++w) Z += ssm[w][lane] * __expf(smx[w][lane] - M);
    const size_t idx = (((size_t)b * H + h) * T + j0 + lane) * 2;
    cs[idx] = M; cs[idx + 1] = Z;
  }
}

// Pass 2: out = P @ V with P = exp(S*c - m_j)/Z_j ; fused residual:  x2 = out + x.
__global__ void k_attn_out(const _Float16* __restrict__ qkv, const float* __restrict__ cs,
                           const float* __restrict__ x, float* __restrict__ x2) {
  __shared__ _Float16 lds[8][16 * 32];
  const int lane = threadIdx.x & 31, wave = threadIdx.x >> 5;
  const int n = lane & 15, half = lane >> 4;
  const int i0 = (blockIdx.x * 8 + wave) * 16;
  const int h = blockIdx.y, b = blockIdx.z;
  const _Float16* Qb = qkv + (size_t)b * T * E3 + h * Dh;
  const _Float16* Kb = Qb + E;
  const _Float16* Vb = Qb + 2 * E;
  const float* csb = cs + ((size_t)b * H + h) * T * 2;
  _Float16* P = &lds[wave][0];

  v8f acc[4] = {};
  for (int jp = 0; jp <= i0; jp += 32) {
#pragma unroll
    for (int sub = 0; sub < 2; ++sub) {
      const int j0 = jp + sub * 16;
      if (j0 <= i0) {
        const v8f c  = s_tile(Qb, Kb, i0, j0);
        const float mj = csb[(j0 + n) * 2];
        const float rz = 1.0f / csb[(j0 + n) * 2 + 1];    // Z >= 1 always
#pragma unroll
        for (int r = 0; r < 8; ++r) {
          const int m = r + 8 * half;
          const float p = ((i0 + m) >= (j0 + n)) ? __expf(c[r] * SCALE - mj) * rz : 0.f;
          P[m * 32 + sub * 16 + n] = (_Float16)p;
        }
      } else {
#pragma unroll
        for (int r = 0; r < 8; ++r) P[(r + 8 * half) * 32 + sub * 16 + n] = (_Float16)0.f;
      }
    }
    __asm__ volatile("s_wait_dscnt 0" ::: "memory");      // C-layout -> A-layout via LDS
    const v16h a = load_a_rm(P, 32);
#pragma unroll
    for (int t = 0; t < 4; ++t) {
      const v16h bf = load_b_nn(Vb + (size_t)jp * E3 + t * 16, E3);
      acc[t] = WMMA_F16(a, bf, acc[t]);
    }
  }
#pragma unroll
  for (int t = 0; t < 4; ++t) {
#pragma unroll
    for (int r = 0; r < 8; ++r) {
      const int m = r + 8 * half;
      const int col = h * Dh + 16 * t + n;
      const size_t gi = ((size_t)b * T + i0 + m) * E + col;
      x2[gi] = acc[t][r] + x[gi];
    }
  }
}

// ---------------- host-side orchestration ----------------
extern "C" void kernel_launch(void* const* d_in, const int* in_sizes, int n_in,
                              void* d_out, int out_size, void* d_ws, size_t ws_size,
                              hipStream_t stream) {
  const float* x  = (const float*)d_in[0];
  const float* ws = (const float*)d_in[1];
  const float* g1 = (const float*)d_in[2];
  const float* g2 = (const float*)d_in[3];
  const float* w1 = (const float*)d_in[4];
  const float* b1 = (const float*)d_in[5];
  const float* w2 = (const float*)d_in[6];
  const float* b2 = (const float*)d_in[7];
  float* out = (float*)d_out;

  // scratch layout (bytes); FFN hidden (64 MiB) aliases h_f16+qkv_f16 (dead by then)
  char* wsb = (char*)d_ws;
  constexpr size_t MiB = 1024 * 1024;
  _Float16* hf16  = (_Float16*)(wsb + 0);            // 16 MiB  [MT x E]
  _Float16* qkv   = (_Float16*)(wsb + 16 * MiB);     // 48 MiB  [MT x 3E]
  _Float16* ff1   = (_Float16*)(wsb + 0);            // 64 MiB  [MT x 4E] (alias)
  float*    cstat = (float*)   (wsb + 64 * MiB);     //  1 MiB  [B*H*T x 2]
  float*    x2    = (float*)   (wsb + 65 * MiB);     // 32 MiB  [MT x E]
  _Float16* h2    = (_Float16*)(wsb + 97 * MiB);     // 16 MiB  [MT x E]
  _Float16* wsT   = (_Float16*)(wsb + 113 * MiB);    //  6 MiB  [3E x E]
  _Float16* w1T   = (_Float16*)(wsb + 119 * MiB);    //  8 MiB  [4E x E]
  _Float16* w2T   = (_Float16*)(wsb + 127 * MiB);    //  8 MiB  [E x 4E]

  // 1) weights -> f16, transposed to [N x K] for contiguous WMMA-B loads
  k_cvt_transpose<<<(E * E3 + 255) / 256, 256, 0, stream>>>(ws, wsT, E,  E3);
  k_cvt_transpose<<<(E * E4 + 255) / 256, 256, 0, stream>>>(w1, w1T, E,  E4);
  k_cvt_transpose<<<(E * E4 + 255) / 256, 256, 0, stream>>>(w2, w2T, E4, E);

  // 2) LN1 -> f16
  k_layernorm_f16<<<MT, 256, 0, stream>>>(x, g1, hf16);

  // 3) QKV = h @ ws           (M=8192, N=3072, K=1024) -> f16
  k_gemm_f16<<<dim3(E3 / 64, MT / 128), 128, 0, stream>>>(
      hf16, wsT, nullptr, nullptr, nullptr, qkv, MT, E3, E, 0);

  // 4) attention pass 1: column softmax stats (quirk: softmax over query axis)
  k_attn_colstats<<<dim3(T / 16, H, Bc), 256, 0, stream>>>(qkv, cstat);

  // 5) attention pass 2: x2 = attn@V + x
  k_attn_out<<<dim3(T / 128, H, Bc), 256, 0, stream>>>(qkv, cstat, x, x2);

  // 6) LN2 -> f16
  k_layernorm_f16<<<MT, 256, 0, stream>>>(x2, g2, h2);

  // 7) ff1 = relu(h2 @ w1 + b1)  (N=4096) -> f16
  k_gemm_f16<<<dim3(E4 / 64, MT / 128), 128, 0, stream>>>(
      h2, w1T, b1, nullptr, nullptr, ff1, MT, E4, E, 1);

  // 8) out = ff1 @ w2 + b2 + x2  (N=1024, K=4096) -> f32
  k_gemm_f16<<<dim3(E / 64, MT / 128), 128, 0, stream>>>(
      ff1, w2T, b2, x2, out, nullptr, MT, E, E4, 0);
}
